// fVSRN_88252987998739
// MI455X (gfx1250) — matrix operand order repaired and checked by reading, not compiled, not run on hardware
//
#include <hip/hip_runtime.h>
#include <hip/hip_bf16.h>

typedef __attribute__((ext_vector_type(16))) _Float16 v16h;
typedef __attribute__((ext_vector_type(8)))  float    v8f;
typedef __attribute__((ext_vector_type(8)))  _Float16 h8;

#define GRID   64
#define NVOX   (GRID*GRID*GRID)
#define NFEAT  64
#define HID    128
#define LPE    6
#define NEG_SLOPE 0.01f
#define PTS_PER_BLOCK 256
#define ASTRIDE 136                 // padded halves per row (bank-conflict avoidance)

// d_ws layout (bytes)
#define WS_WT    0                          // _Float16[4][128][128], wt[l][n][k]
#define WS_BIAS  (4*HID*HID*2)              // float[4][128]
#define WS_W4    (WS_BIAS + 4*HID*4)        // _Float16[128]
#define WS_B4    (WS_W4 + HID*2)            // float
#define WS_GRIDT 133632                     // _Float16[64][64][64][64] voxel-major, 33.5 MB
#define WS_NEED  ((size_t)WS_GRIDT + (size_t)NVOX*NFEAT*2)

union ABFrag { v16h v; uint4 q[2]; };

// Input feature order for layer 0: [feats(0..63) | pe(64..99) | zero pad(100..127)]
// -> wt[0][n][k] uses permuted rows of w0 (reference order is [pe(36) | feats(64)]).
__global__ void fvsrn_prep_weights(
    const float* __restrict__ w0, const float* __restrict__ b0,
    const float* __restrict__ w1, const float* __restrict__ b1,
    const float* __restrict__ w2, const float* __restrict__ b2,
    const float* __restrict__ w3, const float* __restrict__ b3,
    const float* __restrict__ w4, const float* __restrict__ b4,
    unsigned char* __restrict__ ws)
{
    _Float16* wt  = (_Float16*)(ws + WS_WT);
    float*    bia = (float*)   (ws + WS_BIAS);
    _Float16* w4h = (_Float16*)(ws + WS_W4);
    float*    b4o = (float*)   (ws + WS_B4);

    int idx = blockIdx.x * blockDim.x + threadIdx.x;
    int total = 4 * HID * HID;
    for (int i = idx; i < total; i += gridDim.x * blockDim.x) {
        int l = i >> 14;            // / (128*128)
        int n = (i >> 7) & 127;
        int k = i & 127;
        float v;
        if (l == 0) {
            if (k < NFEAT)            v = w0[(36 + k) * HID + n];   // feats
            else if (k < NFEAT + 36)  v = w0[(k - NFEAT) * HID + n];// pe
            else                      v = 0.0f;                     // pad
        } else {
            const float* w = (l == 1) ? w1 : (l == 2) ? w2 : w3;
            v = w[k * HID + n];
        }
        wt[i] = (_Float16)v;
    }
    if (idx < HID) {
        bia[0 * HID + idx] = b0[idx];
        bia[1 * HID + idx] = b1[idx];
        bia[2 * HID + idx] = b2[idx];
        bia[3 * HID + idx] = b3[idx];
        w4h[idx] = (_Float16)w4[idx];
    }
    if (idx == 0) b4o[0] = b4[0];
}

// Transpose feature grid [F][vox] -> voxel-major f16 [vox][F] via LDS tiles.
__global__ __launch_bounds__(256)
void fvsrn_prep_grid(const float* __restrict__ fg, unsigned char* __restrict__ ws)
{
    __shared__ _Float16 t[NFEAT][65];
    _Float16* out = (_Float16*)(ws + WS_GRIDT);
    const int v0  = blockIdx.x * 64;     // voxel tile base
    const int tid = threadIdx.x;

    #pragma unroll
    for (int it = 0; it < 16; ++it) {
        int f  = it * 4 + (tid >> 6);
        int vx = tid & 63;
        t[f][vx] = (_Float16)fg[f * NVOX + v0 + vx];
    }
    __syncthreads();

    const int v  = tid >> 2;
    const int f0 = (tid & 3) * 16;
    _Float16* op = out + (size_t)(v0 + v) * NFEAT + f0;
    #pragma unroll
    for (int j = 0; j < 16; ++j) op[j] = t[f0 + j][v];
}

__global__ __launch_bounds__(PTS_PER_BLOCK)
void fvsrn_mlp_kernel(const float* __restrict__ x,
                      const float* __restrict__ fgrid,
                      const unsigned char* __restrict__ ws,
                      float* __restrict__ out,
                      int useT)
{
    __shared__ _Float16 wts[4][HID][ASTRIDE];          // 139264 B
    __shared__ _Float16 act[PTS_PER_BLOCK][ASTRIDE];   //  69632 B
    __shared__ float    bias_s[4][HID];                //   2048 B
    __shared__ _Float16 w4_s[HID];
    __shared__ float    b4_s;

    const int tid  = threadIdx.x;
    const int lane = tid & 31;
    const int wave = tid >> 5;

    // ---- stage packed weights into LDS via async global->LDS (ASYNCcnt path) ----
    {
        const uint4* src = (const uint4*)(ws + WS_WT);   // 4*128*16 uint4 chunks
        for (int c = tid; c < 4 * HID * 16; c += PTS_PER_BLOCK) {
            int l = c >> 11;
            int n = (c >> 4) & 127;
            int q = c & 15;
            unsigned lds_off = (unsigned)(size_t)&wts[l][n][q * 8];
            unsigned long long ga = (unsigned long long)(src + c);
            asm volatile("global_load_async_to_lds_b128 %0, %1, off"
                         :: "v"(lds_off), "v"(ga) : "memory");
        }
        const float* bsrc = (const float*)(ws + WS_BIAS);
        for (int i = tid; i < 4 * HID; i += PTS_PER_BLOCK)
            bias_s[i >> 7][i & 127] = bsrc[i];
        const _Float16* w4src = (const _Float16*)(ws + WS_W4);
        if (tid < HID) w4_s[tid] = w4src[tid];
        if (tid == 0)  b4_s = *(const float*)(ws + WS_B4);
        asm volatile("s_wait_asynccnt 0" ::: "memory");
    }

    // ---- per-point: trilinear grid sample (feats k=0..63) + PE (k=64..99) ----
    {
        const int p = blockIdx.x * PTS_PER_BLOCK + tid;
        const float px = x[p * 3 + 0];
        const float py = x[p * 3 + 1];
        const float pz = x[p * 3 + 2];

        float fx = (px + 1.0f) * 0.5f * (GRID - 1);
        float fy = (py + 1.0f) * 0.5f * (GRID - 1);
        float fz = (pz + 1.0f) * 0.5f * (GRID - 1);
        float x0f = floorf(fx), y0f = floorf(fy), z0f = floorf(fz);
        float wx = fx - x0f, wy = fy - y0f, wz = fz - z0f;
        int x0 = min(max((int)x0f, 0), GRID - 1);
        int y0 = min(max((int)y0f, 0), GRID - 1);
        int z0 = min(max((int)z0f, 0), GRID - 1);
        int x1 = min(x0 + 1, GRID - 1);
        int y1 = min(y0 + 1, GRID - 1);
        int z1 = min(z0 + 1, GRID - 1);

        int off[8];
        off[0] = z0 * 4096 + y0 * 64 + x0;
        off[1] = z0 * 4096 + y0 * 64 + x1;
        off[2] = z0 * 4096 + y1 * 64 + x0;
        off[3] = z0 * 4096 + y1 * 64 + x1;
        off[4] = z1 * 4096 + y0 * 64 + x0;
        off[5] = z1 * 4096 + y0 * 64 + x1;
        off[6] = z1 * 4096 + y1 * 64 + x0;
        off[7] = z1 * 4096 + y1 * 64 + x1;
        float cw[8];
        cw[0] = (1.f - wz) * (1.f - wy) * (1.f - wx);
        cw[1] = (1.f - wz) * (1.f - wy) * wx;
        cw[2] = (1.f - wz) * wy * (1.f - wx);
        cw[3] = (1.f - wz) * wy * wx;
        cw[4] = wz * (1.f - wy) * (1.f - wx);
        cw[5] = wz * (1.f - wy) * wx;
        cw[6] = wz * wy * (1.f - wx);
        cw[7] = wz * wy * wx;

        if (useT) {
            // fast path: voxel-major f16 grid, one 128B cacheline per corner
            const _Float16* gt = (const _Float16*)(ws + WS_GRIDT);
            h8 acc[8];
            #pragma unroll
            for (int j = 0; j < 8; ++j) acc[j] = (h8)(_Float16)0.0f;
            #pragma unroll
            for (int cn = 0; cn < 8; ++cn) {
                const h8* gp = (const h8*)(gt + (size_t)off[cn] * NFEAT);
                _Float16 wc = (_Float16)cw[cn];
                #pragma unroll
                for (int j = 0; j < 8; ++j) acc[j] += gp[j] * wc;   // v_pk_fma_f16
            }
            #pragma unroll
            for (int j = 0; j < 8; ++j) *(h8*)&act[tid][j * 8] = acc[j];
        } else {
            // fallback: original fp32 feature-major grid
            for (int f = 0; f < NFEAT; ++f) {
                const float* g = fgrid + f * NVOX;
                float v = cw[0]*g[off[0]] + cw[1]*g[off[1]] + cw[2]*g[off[2]] + cw[3]*g[off[3]]
                        + cw[4]*g[off[4]] + cw[5]*g[off[5]] + cw[6]*g[off[6]] + cw[7]*g[off[7]];
                act[tid][f] = (_Float16)v;
            }
        }

        float freq = 3.14159265358979323846f;
        #pragma unroll
        for (int l = 0; l < LPE; ++l) {
            float ax = px * freq, ay = py * freq, az = pz * freq;
            act[tid][NFEAT + l * 3 + 0]      = (_Float16)__sinf(ax);
            act[tid][NFEAT + l * 3 + 1]      = (_Float16)__sinf(ay);
            act[tid][NFEAT + l * 3 + 2]      = (_Float16)__sinf(az);
            act[tid][NFEAT + 18 + l * 3 + 0] = (_Float16)__cosf(ax);
            act[tid][NFEAT + 18 + l * 3 + 1] = (_Float16)__cosf(ay);
            act[tid][NFEAT + 18 + l * 3 + 2] = (_Float16)__cosf(az);
            freq *= 2.0f;
        }
        #pragma unroll
        for (int k = NFEAT + 36; k < HID; ++k) act[tid][k] = (_Float16)0.0f;
    }
    __syncthreads();

    // ---- 4 hidden layers, 32x128 f16 GEMM tile per wave via WMMA ----
    const int rbase   = wave * 32;
    const int laneLo  = lane & 15;
    const int laneHiA = (lane >> 4) << 3;   // +8 halves (A) / +8 rows (D)
    const int laneHiB = (lane >> 4) << 4;   // +16 halves (unused; B uses laneHiA)

    #pragma unroll 1
    for (int l = 0; l < 4; ++l) {
        ABFrag a[2][4];
        #pragma unroll
        for (int mt = 0; mt < 2; ++mt) {
            #pragma unroll
            for (int kt = 0; kt < 4; ++kt) {
                const _Float16* rp = &act[rbase + mt * 16 + laneLo][kt * 32 + laneHiA];
                a[mt][kt].q[0] = *(const uint4*)(rp);
                a[mt][kt].q[1] = *(const uint4*)(rp + 16);
            }
        }
        #pragma unroll
        for (int nt = 0; nt < 8; ++nt) {
            const int col = nt * 16 + laneLo;
            v8f acc0 = {};    // inline-0 C on first WMMA, bias added in epilogue
            v8f acc1 = {};
            #pragma unroll
            for (int kt = 0; kt < 4; ++kt) {
                ABFrag b;
                const _Float16* bp = &wts[l][col][kt * 32 + laneHiA];
                b.q[0] = *(const uint4*)(bp);
                b.q[1] = *(const uint4*)(bp + 16);
                acc0 = __builtin_amdgcn_wmma_f32_16x16x32_f16(
                    false, a[0][kt].v, false, b.v, (short)0, acc0, false, false);
                acc1 = __builtin_amdgcn_wmma_f32_16x16x32_f16(
                    false, a[1][kt].v, false, b.v, (short)0, acc1, false, false);
            }
            const float bval = bias_s[l][col];
            #pragma unroll
            for (int i = 0; i < 8; ++i) {
                float h0 = acc0[i] + bval; h0 = (h0 >= 0.0f) ? h0 : NEG_SLOPE * h0;
                float h1 = acc1[i] + bval; h1 = (h1 >= 0.0f) ? h1 : NEG_SLOPE * h1;
                act[rbase + i + laneHiA][col]      = (_Float16)h0;
                act[rbase + 16 + i + laneHiA][col] = (_Float16)h1;
            }
        }
        // no barrier: each wave touches only its own 32 rows; LDS is in-order per wave
    }

    // ---- final 128 -> 1 layer: vectorized per-lane dot product ----
    {
        const int plocal = rbase + lane;
        const h8* ar8 = (const h8*)&act[plocal][0];
        const h8* w48 = (const h8*)&w4_s[0];
        float sum = b4_s;
        #pragma unroll
        for (int j = 0; j < 16; ++j) {
            h8 pa = ar8[j];
            h8 pw = w48[j];
            #pragma unroll
            for (int e = 0; e < 8; ++e)
                sum += (float)pa[e] * (float)pw[e];
        }
        out[blockIdx.x * PTS_PER_BLOCK + plocal] = sum;
    }
}

extern "C" void kernel_launch(void* const* d_in, const int* in_sizes, int n_in,
                              void* d_out, int out_size, void* d_ws, size_t ws_size,
                              hipStream_t stream) {
    const float* x  = (const float*)d_in[0];
    const float* fg = (const float*)d_in[1];
    const float* w0 = (const float*)d_in[2];  const float* b0 = (const float*)d_in[3];
    const float* w1 = (const float*)d_in[4];  const float* b1 = (const float*)d_in[5];
    const float* w2 = (const float*)d_in[6];  const float* b2 = (const float*)d_in[7];
    const float* w3 = (const float*)d_in[8];  const float* b3 = (const float*)d_in[9];
    const float* w4 = (const float*)d_in[10]; const float* b4 = (const float*)d_in[11];

    const int useT = (ws_size >= WS_NEED) ? 1 : 0;

    fvsrn_prep_weights<<<64, 256, 0, stream>>>(w0, b0, w1, b1, w2, b2, w3, b3, w4, b4,
                                               (unsigned char*)d_ws);
    if (useT) {
        fvsrn_prep_grid<<<NVOX / 64, 256, 0, stream>>>(fg, (unsigned char*)d_ws);
    }

    const int nblocks = out_size / PTS_PER_BLOCK;   // 1048576 / 256 = 4096
    fvsrn_mlp_kernel<<<nblocks, PTS_PER_BLOCK, 0, stream>>>(
        x, fg, (const unsigned char*)d_ws, (float*)d_out, useT);
}
